// CausalSelfAttention_51556787421660
// MI455X (gfx1250) — matrix operand order repaired and checked by reading, not compiled, not run on hardware
//
#include <hip/hip_runtime.h>

// CausalSelfAttention B=8, T=2048, C=256, H=8, D=32 for gfx1250 (MI455X).
// Pipeline: weight f32->f16 convert -> qkv GEMM (WMMA) ->
//           flash attention (WMMA + async-to-LDS K staging) -> proj GEMM (WMMA).

typedef __attribute__((ext_vector_type(16))) _Float16 v16h;
typedef __attribute__((ext_vector_type(8)))  _Float16 v8h;
typedef __attribute__((ext_vector_type(8)))  float    v8f;

union AFrag { v16h v; v8h h2[2]; _Float16 h[16]; };
union CFrag { v8f  v; float f[8]; };

#define NB  8
#define NT  2048
#define NC  256
#define NH  8
#define ND  32

// ---------------------------------------------------------------------------
// Kernel 0: one-shot weight conversion f32 -> f16 (keeps GEMM B-frags cvt-free).
// ---------------------------------------------------------------------------
__global__ __launch_bounds__(256) void cvt_weights(const float* __restrict__ wa,
                                                   const float* __restrict__ wp,
                                                   _Float16* __restrict__ wa16,
                                                   _Float16* __restrict__ wp16) {
  const int i = blockIdx.x * 256 + threadIdx.x;          // grid covers 196608
  if (i < 196608) wa16[i] = (_Float16)wa[i];
  if (i < 65536)  wp16[i] = (_Float16)wp[i];
}

// ---------------------------------------------------------------------------
// Kernel 1: qkv = x[16384,256] @ w_attn[256,768] (w pre-converted to f16),
// scattered to q/k/v f16 arrays laid out [B,H,T,D]. One wave per 16x16 tile,
// K=256 in 8 wmma_f32_16x16x32_f16 steps.
// ---------------------------------------------------------------------------
__global__ __launch_bounds__(256) void qkv_gemm(const float* __restrict__ x,
                                                const _Float16* __restrict__ w16,
                                                _Float16* __restrict__ q,
                                                _Float16* __restrict__ k,
                                                _Float16* __restrict__ v) {
  const int wid  = threadIdx.x >> 5;
  const int lane = threadIdx.x & 31;
  const int tile = blockIdx.x * 8 + wid;           // 49152 tiles
  const int m0 = (tile / 48) * 16;                 // token tile
  const int n0 = (tile % 48) * 16;                 // output-col tile (0..767)
  const int h2 = lane >> 4;
  const int mrow = m0 + (lane & 15);

  CFrag acc = {};
#pragma unroll
  for (int kk = 0; kk < 256; kk += 32) {
    // A fragment (f32 activations -> f16): e<8 -> K=kk+8*h2+e, e>=8 -> +16.
    AFrag a;
    const float* xr = x + (size_t)mrow * 256 + kk + 8 * h2;
    float4 fa = *(const float4*)(xr + 0);
    float4 fb = *(const float4*)(xr + 4);
    float4 fc = *(const float4*)(xr + 16);
    float4 fd = *(const float4*)(xr + 20);
    a.h[0]=(_Float16)fa.x; a.h[1]=(_Float16)fa.y; a.h[2]=(_Float16)fa.z; a.h[3]=(_Float16)fa.w;
    a.h[4]=(_Float16)fb.x; a.h[5]=(_Float16)fb.y; a.h[6]=(_Float16)fb.z; a.h[7]=(_Float16)fb.w;
    a.h[8]=(_Float16)fc.x; a.h[9]=(_Float16)fc.y; a.h[10]=(_Float16)fc.z; a.h[11]=(_Float16)fc.w;
    a.h[12]=(_Float16)fd.x; a.h[13]=(_Float16)fd.y; a.h[14]=(_Float16)fd.z; a.h[15]=(_Float16)fd.w;

    // B fragment: lane = N column, elements = K rows in table order (pure f16 loads).
    AFrag b;
#pragma unroll
    for (int e = 0; e < 16; ++e)
      b.h[e] = w16[(size_t)(kk + (e & 7) + 8 * h2 + 16 * (e >> 3)) * 768 + n0 + (lane & 15)];

    acc.v = __builtin_amdgcn_wmma_f32_16x16x32_f16(false, a.v, false, b.v,
                                                   (short)0, acc.v, false, false);
  }

  const int mat = n0 / 256;                        // 0=q,1=k,2=v
  _Float16* dst = (mat == 0) ? q : (mat == 1) ? k : v;
  const int col = (n0 % 256) + (lane & 15);
  const int hh = col >> 5, dd = col & 31;
#pragma unroll
  for (int r = 0; r < 8; ++r) {
    const int m = m0 + r + 8 * h2;
    const int bb = m >> 11, tt = m & 2047;
    dst[(((size_t)bb * NH + hh) * NT + tt) * ND + dd] = (_Float16)acc.f[r];
  }
}

// ---------------------------------------------------------------------------
// Kernel 2: causal flash attention. One wave per (b,h, 32-query tile).
// Per 32-key block: async-stage K into LDS, 4 WMMAs for S, online softmax,
// P relayout through LDS, 4 WMMAs for O += P V.
// ---------------------------------------------------------------------------
__global__ __launch_bounds__(32) void attn_kernel(const _Float16* __restrict__ q,
                                                  const _Float16* __restrict__ k,
                                                  const _Float16* __restrict__ v,
                                                  _Float16* __restrict__ y) {
  __shared__ _Float16 ldsK[32 * 32];   // [key][d] staging (async filled)
  __shared__ _Float16 ldsP[32 * 32];   // [m][key] P relayout

  const int lane = threadIdx.x & 31;
  const int q0 = blockIdx.x * 32;
  const int bh = blockIdx.y;
  const int b = bh >> 3, h = bh & 7;
  const int h2 = lane >> 4;
  const int nn = lane & 15;
  const float scale = 0.17677669529663687f;        // 1/sqrt(32)

  const _Float16* qb = q + (size_t)bh * NT * ND;
  const _Float16* kb = k + (size_t)bh * NT * ND;
  const _Float16* vb = v + (size_t)bh * NT * ND;

  // Q fragments (A layout): rows q0+qt*16+nn.
  AFrag qa[2];
#pragma unroll
  for (int qt = 0; qt < 2; ++qt) {
    const _Float16* qr = qb + (size_t)(q0 + qt * 16 + nn) * ND;
    qa[qt].h2[0] = *(const v8h*)(qr + 8 * h2);
    qa[qt].h2[1] = *(const v8h*)(qr + 16 + 8 * h2);
  }

  float m_i[2][8], l_i[2][8];
#pragma unroll
  for (int qt = 0; qt < 2; ++qt)
#pragma unroll
    for (int r = 0; r < 8; ++r) { m_i[qt][r] = -1e30f; l_i[qt][r] = 0.0f; }
  CFrag O[2][2] = {};

  // LDS byte offset of this lane's K staging row (flat addr low 32 bits = LDS addr).
  const uint32_t ldsKrow = (uint32_t)(size_t)(ldsK) + (uint32_t)lane * 64u;

  for (int k0 = 0; k0 < q0 + 32; k0 += 32) {
    // ---- Async stage K block [32 keys x 32 dims] directly into LDS --------
    {
      const _Float16* krow = kb + (size_t)(k0 + lane) * ND;   // 64B per key row
      asm volatile(
          "global_load_async_to_lds_b128 %0, %1, off\n\t"
          "global_load_async_to_lds_b128 %0, %1, off offset:16\n\t"
          "global_load_async_to_lds_b128 %0, %1, off offset:32\n\t"
          "global_load_async_to_lds_b128 %0, %1, off offset:48\n\t"
          "s_wait_asynccnt 0x0"
          :: "v"(ldsKrow), "v"(krow) : "memory");
    }
    __syncthreads();

    // Kt B fragments: lane = key, elements = d in table order.
    AFrag bk0, bk1;
    bk0.h2[0] = *(const v8h*)&ldsK[(size_t)nn * 32 + 8 * h2];
    bk0.h2[1] = *(const v8h*)&ldsK[(size_t)nn * 32 + 16 + 8 * h2];
    bk1.h2[0] = *(const v8h*)&ldsK[(size_t)(nn + 16) * 32 + 8 * h2];
    bk1.h2[1] = *(const v8h*)&ldsK[(size_t)(nn + 16) * 32 + 16 + 8 * h2];

    // S tiles: S[qt][kt] = Q[qt] x Kt[kt].
    CFrag S[2][2];
#pragma unroll
    for (int qt = 0; qt < 2; ++qt) {
      v8f z = {};
      S[qt][0].v = __builtin_amdgcn_wmma_f32_16x16x32_f16(false, qa[qt].v, false, bk0.v,
                                                          (short)0, z, false, false);
      S[qt][1].v = __builtin_amdgcn_wmma_f32_16x16x32_f16(false, qa[qt].v, false, bk1.v,
                                                          (short)0, z, false, false);
    }

    // Online softmax per row (row M = qt*16 + r + 8*h2, col = nn).
#pragma unroll
    for (int qt = 0; qt < 2; ++qt) {
#pragma unroll
      for (int r = 0; r < 8; ++r) {
        const int mg = q0 + qt * 16 + r + 8 * h2;
        const int kg0 = k0 + nn, kg1 = kg0 + 16;
        float s0 = (kg0 <= mg) ? S[qt][0].f[r] * scale : -1e30f;
        float s1 = (kg1 <= mg) ? S[qt][1].f[r] * scale : -1e30f;

        float mx = fmaxf(s0, s1);
#pragma unroll
        for (int off = 8; off >= 1; off >>= 1)
          mx = fmaxf(mx, __shfl_xor(mx, off, 16));
        const float mnew = fmaxf(m_i[qt][r], mx);
        const float alpha = __expf(m_i[qt][r] - mnew);
        const float p0 = __expf(s0 - mnew);
        const float p1 = __expf(s1 - mnew);
        float rs = p0 + p1;
#pragma unroll
        for (int off = 8; off >= 1; off >>= 1)
          rs += __shfl_xor(rs, off, 16);
        l_i[qt][r] = l_i[qt][r] * alpha + rs;
        m_i[qt][r] = mnew;
        O[qt][0].f[r] *= alpha;
        O[qt][1].f[r] *= alpha;
        ldsP[(size_t)(qt * 16 + r + 8 * h2) * 32 + nn]      = (_Float16)p0;
        ldsP[(size_t)(qt * 16 + r + 8 * h2) * 32 + 16 + nn] = (_Float16)p1;
      }
    }
    __syncthreads();

    // P as A fragments (row = qt*16 + nn, K = key).
    AFrag pa[2];
#pragma unroll
    for (int qt = 0; qt < 2; ++qt) {
      pa[qt].h2[0] = *(const v8h*)&ldsP[(size_t)(qt * 16 + nn) * 32 + 8 * h2];
      pa[qt].h2[1] = *(const v8h*)&ldsP[(size_t)(qt * 16 + nn) * 32 + 16 + 8 * h2];
    }

    // V B fragments: lane = dim, elements = key in table order.
    AFrag bv0, bv1;
#pragma unroll
    for (int e = 0; e < 16; ++e) {
      const _Float16* vr = vb + (size_t)(k0 + (e & 7) + 8 * h2 + 16 * (e >> 3)) * ND;
      bv0.h[e] = vr[nn];
      bv1.h[e] = vr[16 + nn];
    }

#pragma unroll
    for (int qt = 0; qt < 2; ++qt) {
      O[qt][0].v = __builtin_amdgcn_wmma_f32_16x16x32_f16(false, pa[qt].v, false, bv0.v,
                                                          (short)0, O[qt][0].v, false, false);
      O[qt][1].v = __builtin_amdgcn_wmma_f32_16x16x32_f16(false, pa[qt].v, false, bv1.v,
                                                          (short)0, O[qt][1].v, false, false);
    }
    __syncthreads();
  }

  // Write y[b][t][h*32 + d] (f16 for proj input).
#pragma unroll
  for (int qt = 0; qt < 2; ++qt) {
#pragma unroll
    for (int r = 0; r < 8; ++r) {
      const int t = q0 + qt * 16 + r + 8 * h2;
      const float inv = 1.0f / l_i[qt][r];
      _Float16* yr = y + ((size_t)b * NT + t) * NC + h * ND;
      yr[nn]      = (_Float16)(O[qt][0].f[r] * inv);
      yr[16 + nn] = (_Float16)(O[qt][1].f[r] * inv);
    }
  }
}

// ---------------------------------------------------------------------------
// Kernel 3: out = y[16384,256](f16) @ w_proj16[256,256](f16), f32 output.
// ---------------------------------------------------------------------------
__global__ __launch_bounds__(256) void proj_gemm(const _Float16* __restrict__ y,
                                                 const _Float16* __restrict__ w16,
                                                 float* __restrict__ out) {
  const int wid  = threadIdx.x >> 5;
  const int lane = threadIdx.x & 31;
  const int tile = blockIdx.x * 8 + wid;           // 16384 tiles
  const int m0 = (tile / 16) * 16;
  const int n0 = (tile % 16) * 16;
  const int h2 = lane >> 4;
  const int mrow = m0 + (lane & 15);

  CFrag acc = {};
#pragma unroll
  for (int kk = 0; kk < 256; kk += 32) {
    AFrag a;
    const _Float16* yr = y + (size_t)mrow * 256 + kk + 8 * h2;
    a.h2[0] = *(const v8h*)(yr + 0);
    a.h2[1] = *(const v8h*)(yr + 16);

    AFrag b;
#pragma unroll
    for (int e = 0; e < 16; ++e)
      b.h[e] = w16[(size_t)(kk + (e & 7) + 8 * h2 + 16 * (e >> 3)) * 256 + n0 + (lane & 15)];

    acc.v = __builtin_amdgcn_wmma_f32_16x16x32_f16(false, a.v, false, b.v,
                                                   (short)0, acc.v, false, false);
  }

#pragma unroll
  for (int r = 0; r < 8; ++r) {
    out[(size_t)(m0 + r + 8 * h2) * 256 + n0 + (lane & 15)] = acc.f[r];
  }
}

// ---------------------------------------------------------------------------
extern "C" void kernel_launch(void* const* d_in, const int* in_sizes, int n_in,
                              void* d_out, int out_size, void* d_ws, size_t ws_size,
                              hipStream_t stream) {
  const float* x      = (const float*)d_in[0];   // [8,2048,256]
  const float* w_attn = (const float*)d_in[1];   // [256,768]
  const float* w_proj = (const float*)d_in[2];   // [256,256]
  float* out = (float*)d_out;                    // [8,2048,256]

  // Workspace (f16 elements): q,k,v [B,H,T,D] + y [B,T,C] + converted weights.
  _Float16* ws   = (_Float16*)d_ws;
  _Float16* q    = ws;
  _Float16* k    = ws + (size_t)4194304;
  _Float16* v    = ws + (size_t)8388608;
  _Float16* y    = ws + (size_t)12582912;
  _Float16* wa16 = ws + (size_t)16777216;        // 196608 halfs
  _Float16* wp16 = ws + (size_t)16973824;        // 65536 halfs

  cvt_weights<<<768, 256, 0, stream>>>(w_attn, w_proj, wa16, wp16);
  qkv_gemm<<<6144, 256, 0, stream>>>(x, wa16, q, k, v);
  attn_kernel<<<dim3(64, 64), 32, 0, stream>>>(q, k, v, y);
  proj_gemm<<<2048, 256, 0, stream>>>(y, wp16, out);
}
